// CFCEncoder_17781164605877
// MI455X (gfx1250) — matrix operand order repaired and checked by reading, not compiled, not run on hardware
//
#include <hip/hip_runtime.h>
#include <hip/hip_bf16.h>
#include <math.h>

// ---------------------------------------------------------------------------
// CfC (liquid NN) encoder for MI455X / gfx1250.
// Batch-parallel over the recurrence: each workgroup owns 16 batch rows for
// the entire K=64 sequence; all GEMMs via v_wmma_f32_16x16x32_bf16.
// Weights packed once (bf16, mask folded, ta+tb merged) into d_ws.
// ---------------------------------------------------------------------------

typedef __attribute__((ext_vector_type(16))) __bf16 v16bf;
typedef __attribute__((ext_vector_type(8)))  __bf16 v8bf;
typedef __attribute__((ext_vector_type(8)))  float  v8f;

#define THREADS 512
#define NWAVES  16
#define BT      16        // batch tile (= WMMA M)
#define SEQK    64
#define EBD_    256
#define VFD_    512
#define SENS_   768       // EBD + VFD
#define HID0    512
#define HID1    256
#define HID2    64

struct CfcParams {
  const float* base;            // (B, K, 256)
  const float* vis;             // (B, K, 512)
  const __bf16* w[3];           // packed [s=ff1|ff2|tsum][nt][kt] B-fragments
  const float* b1[3];           // ff1_b
  const float* b2[3];           // ff2_b
  const float* bta[3];          // ta_b
  const float* btb[3];          // tb_b
  float* out;                   // (B, 64)
};

__device__ __forceinline__ float sigm_f(float x) {
  return 1.0f / (1.0f + __expf(-x));
}

// ---------------------------------------------------------------------------
// Weight packer: stacked S=3 matrices (ff1*mask, ff2*mask, ta+tb), bf16,
// laid out in WMMA B-fragment order:
//   p = ((s*NT + nt)*KT + kt)*512 + lane*16 + e
//   n = nt*16 + (lane & 15)
//   k = kt*32 + (lane >> 4)*8 + (e & 7) + (e >> 3)*16   (B[k][n] = W[n][k])
// ---------------------------------------------------------------------------
__global__ void pack_weights(const float* __restrict__ wf1,
                             const float* __restrict__ wf2,
                             const float* __restrict__ wta,
                             const float* __restrict__ wtb,
                             const float* __restrict__ mask,
                             __bf16* __restrict__ dst,
                             int H, int C, int NT, int KT)
{
  size_t total = (size_t)3 * H * C;
  size_t p = (size_t)blockIdx.x * blockDim.x + threadIdx.x;
  if (p >= total) return;
  int e    = (int)(p & 15);
  int lane = (int)((p >> 4) & 31);
  size_t q = p >> 9;
  int kt = (int)(q % KT); q /= KT;
  int nt = (int)(q % NT);
  int s  = (int)(q / NT);
  int n = nt * 16 + (lane & 15);
  int k = kt * 32 + (lane >> 4) * 8 + (e & 7) + (e >> 3) * 16;
  size_t idx = (size_t)n * C + k;
  float v;
  if      (s == 0) v = wf1[idx] * mask[idx];
  else if (s == 1) v = wf2[idx] * mask[idx];
  else             v = wta[idx] + wtb[idx];     // ts==1.0 => gate = sigmoid((ta+tb)x + ...)
  dst[p] = (__bf16)v;
}

// ---------------------------------------------------------------------------
// One CfC layer for the workgroup's 16-row batch tile.
// Input = concat(inA (width WA, KTA k-tiles), inB (width WB, KTB k-tiles)).
// K loop is statically split at the concat boundary: no per-iteration
// region selects in the hot loop. All strides/trip counts compile-time.
// Each wave owns output column tiles nt = wave, wave+16, ... (<= 2 tiles).
// Results held in registers across a barrier, then written to outH (bf16).
// ---------------------------------------------------------------------------
template<int NT, int KTA, int KTB, int WA, int WB, int WH>
__device__ __forceinline__ void layer_proc(
    const __bf16* __restrict__ wpack,
    const __bf16* __restrict__ inA,
    const __bf16* __restrict__ inB,
    const float* __restrict__ pb1, const float* __restrict__ pb2,
    const float* __restrict__ pbta, const float* __restrict__ pbtb,
    __bf16* __restrict__ outH)
{
  constexpr int KT = KTA + KTB;
  const int tid  = threadIdx.x;
  const int wave = tid >> 5;
  const int lane = tid & 31;
  const int m = lane & 15;      // A-matrix row for this lane
  const int g = lane >> 4;      // half-wave selector (K-group / output row group)

  float hreg[2][8];
  int   ntKeep[2];
  int   cnt = 0;

  constexpr size_t sStride = (size_t)NT * KT * 512;  // stride between stacked mats

  // Per-lane base pointers for the A fragment halves (ISA 16-bit A 16x32
  // layout): lane m, K-offsets g*8 and 16+g*8 inside each 32-wide K tile.
  const __bf16* aBaseA = inA + m * WA + g * 8;
  const __bf16* aBaseB = inB + m * WB + g * 8;

  for (int nt = wave; nt < NT && cnt < 2; nt += NWAVES) {
    v8f a0 = {}; v8f a1 = {}; v8f a2 = {};
    const __bf16* p0 = wpack + (size_t)nt * KT * 512 + lane * 16;
    const __bf16* p1 = p0 + sStride;
    const __bf16* p2 = p1 + sStride;

#pragma unroll 4
    for (int kt = 0; kt < KTA; ++kt) {
      const __bf16* arow = aBaseA + kt * 32;
      v8bf alo = *(const v8bf*)(arow);
      v8bf ahi = *(const v8bf*)(arow + 16);
      v16bf af = __builtin_shufflevector(alo, ahi,
                   0,1,2,3,4,5,6,7,8,9,10,11,12,13,14,15);
      v16bf bf0 = *(const v16bf*)(p0 + (size_t)kt * 512);
      v16bf bf1 = *(const v16bf*)(p1 + (size_t)kt * 512);
      v16bf bf2 = *(const v16bf*)(p2 + (size_t)kt * 512);
      a0 = __builtin_amdgcn_wmma_f32_16x16x32_bf16(false, af, false, bf0,
                                                   (short)0, a0, false, false);
      a1 = __builtin_amdgcn_wmma_f32_16x16x32_bf16(false, af, false, bf1,
                                                   (short)0, a1, false, false);
      a2 = __builtin_amdgcn_wmma_f32_16x16x32_bf16(false, af, false, bf2,
                                                   (short)0, a2, false, false);
    }

#pragma unroll 4
    for (int kt = 0; kt < KTB; ++kt) {
      const __bf16* arow = aBaseB + kt * 32;
      v8bf alo = *(const v8bf*)(arow);
      v8bf ahi = *(const v8bf*)(arow + 16);
      v16bf af = __builtin_shufflevector(alo, ahi,
                   0,1,2,3,4,5,6,7,8,9,10,11,12,13,14,15);
      v16bf bf0 = *(const v16bf*)(p0 + (size_t)(KTA + kt) * 512);
      v16bf bf1 = *(const v16bf*)(p1 + (size_t)(KTA + kt) * 512);
      v16bf bf2 = *(const v16bf*)(p2 + (size_t)(KTA + kt) * 512);
      a0 = __builtin_amdgcn_wmma_f32_16x16x32_bf16(false, af, false, bf0,
                                                   (short)0, a0, false, false);
      a1 = __builtin_amdgcn_wmma_f32_16x16x32_bf16(false, af, false, bf1,
                                                   (short)0, a1, false, false);
      a2 = __builtin_amdgcn_wmma_f32_16x16x32_bf16(false, af, false, bf2,
                                                   (short)0, a2, false, false);
    }

    // CfC cell elementwise: h = ff1 + t*(ff2-ff1), t = sigmoid(gate pre-acts)
    int n = nt * 16 + m;
    float vb1 = pb1[n];
    float vb2 = pb2[n];
    float vbt = pbta[n] + pbtb[n];
#pragma unroll
    for (int v = 0; v < 8; ++v) {
      float f1 = tanhf(a0[v] + vb1);
      float f2 = tanhf(a1[v] + vb2);
      float ti = sigm_f(a2[v] + vbt);
      hreg[cnt][v] = f1 + ti * (f2 - f1);
    }
    ntKeep[cnt] = nt;
    ++cnt;
  }

  __syncthreads();   // all recurrent reads of outH done before overwriting
  for (int i = 0; i < cnt; ++i) {
    int n = ntKeep[i] * 16 + m;
#pragma unroll
    for (int v = 0; v < 8; ++v) {
      // C/D layout: lane l, vgpr v -> row (l>>4)*8 + v, col l&15 (within tile)
      outH[(g * 8 + v) * WH + n] = (__bf16)hreg[i][v];
    }
  }
  __syncthreads();   // h fully published for next layer / next timestep
}

// ---------------------------------------------------------------------------
// Main kernel: 32 workgroups x 512 threads. Each WG owns batch rows
// [blockIdx.x*16, +16) for the entire 64-step sequence; no cross-WG sync.
// ---------------------------------------------------------------------------
__global__ __launch_bounds__(THREADS, 1)
void cfc_main(CfcParams P)
{
  __shared__ __align__(16) __bf16 sX [BT * SENS_];   // 24 KB
  __shared__ __align__(16) __bf16 sH0[BT * HID0];    // 16 KB
  __shared__ __align__(16) __bf16 sH1[BT * HID1];    //  8 KB
  __shared__ __align__(16) __bf16 sH2[BT * HID2];    //  2 KB

  const int tid    = threadIdx.x;
  const int wgBase = blockIdx.x * BT;

  for (int i = tid; i < BT * HID0; i += THREADS) sH0[i] = (__bf16)0.0f;
  for (int i = tid; i < BT * HID1; i += THREADS) sH1[i] = (__bf16)0.0f;
  for (int i = tid; i < BT * HID2; i += THREADS) sH2[i] = (__bf16)0.0f;
  __syncthreads();

  for (int t = 0; t < SEQK; ++t) {
    // Stage x_t = concat(base[:,t,:], visual[:,t,:]) into LDS as bf16.
    for (int i = tid; i < BT * SENS_; i += THREADS) {
      int r = i / SENS_;
      int c = i - r * SENS_;
      float v = (c < EBD_)
        ? P.base[((size_t)(wgBase + r) * SEQK + t) * EBD_ + c]
        : P.vis [((size_t)(wgBase + r) * SEQK + t) * VFD_ + (c - EBD_)];
      sX[i] = (__bf16)v;
    }
    __syncthreads();

    // Layer 0: cat = 768(X) + 512(h0), H = 512  (NT=32, KTA=24, KTB=16)
    layer_proc<32, 24, 16, SENS_, HID0, HID0>(
        P.w[0], sX, sH0, P.b1[0], P.b2[0], P.bta[0], P.btb[0], sH0);
    // Layer 1: cat = 512(h0) + 256(h1), H = 256  (NT=16, KTA=16, KTB=8)
    layer_proc<16, 16, 8, HID0, HID1, HID1>(
        P.w[1], sH0, sH1, P.b1[1], P.b2[1], P.bta[1], P.btb[1], sH1);
    // Layer 2: cat = 256(h1) + 64(h2), H = 64    (NT=4, KTA=8, KTB=2)
    layer_proc<4, 8, 2, HID1, HID2, HID2>(
        P.w[2], sH1, sH2, P.b1[2], P.b2[2], P.bta[2], P.btb[2], sH2);
  }

  // Final motor-layer state -> fp32 output.
  for (int i = tid; i < BT * HID2; i += THREADS) {
    int r = i >> 6;
    int n = i & 63;
    P.out[(size_t)(wgBase + r) * HID2 + n] = (float)sH2[i];
  }
}

// ---------------------------------------------------------------------------
// Host launcher
// ---------------------------------------------------------------------------
extern "C" void kernel_launch(void* const* d_in, const int* in_sizes, int n_in,
                              void* d_out, int out_size, void* d_ws, size_t ws_size,
                              hipStream_t stream)
{
  (void)in_sizes; (void)n_in; (void)out_size; (void)ws_size;

  const float* base = (const float*)d_in[0];
  const float* vis  = (const float*)d_in[1];

  // Per-layer param order in d_in: ff1_w, ff1_b, ff2_w, ff2_b,
  //                                ta_w, ta_b, tb_w, tb_b, mask
  const float* W[3][4];
  const float* Bb[3][4];
  const float* Mk[3];
  for (int li = 0; li < 3; ++li) {
    int o = 2 + li * 9;
    W[li][0] = (const float*)d_in[o + 0]; Bb[li][0] = (const float*)d_in[o + 1];
    W[li][1] = (const float*)d_in[o + 2]; Bb[li][1] = (const float*)d_in[o + 3];
    W[li][2] = (const float*)d_in[o + 4]; Bb[li][2] = (const float*)d_in[o + 5];
    W[li][3] = (const float*)d_in[o + 6]; Bb[li][3] = (const float*)d_in[o + 7];
    Mk[li]   = (const float*)d_in[o + 8];
  }

  const int Hs[3] = {HID0, HID1, HID2};
  const int Cs[3] = {SENS_ + HID0, HID0 + HID1, HID1 + HID2};  // 1280, 768, 320

  // Packed bf16 weights live in scratch (~5.2 MB total).
  __bf16* wp[3];
  size_t offElems = 0;
  for (int li = 0; li < 3; ++li) {
    wp[li] = (__bf16*)((char*)d_ws + offElems * sizeof(__bf16));
    offElems += (size_t)3 * Hs[li] * Cs[li];
  }

  for (int li = 0; li < 3; ++li) {
    size_t total = (size_t)3 * Hs[li] * Cs[li];
    int grid = (int)((total + 255) / 256);
    pack_weights<<<grid, 256, 0, stream>>>(W[li][0], W[li][1], W[li][2], W[li][3],
                                           Mk[li], wp[li], Hs[li], Cs[li],
                                           Hs[li] / 16, Cs[li] / 32);
  }

  CfcParams P;
  P.base = base;
  P.vis  = vis;
  P.out  = (float*)d_out;
  for (int li = 0; li < 3; ++li) {
    P.w[li]   = wp[li];
    P.b1[li]  = Bb[li][0];
    P.b2[li]  = Bb[li][1];
    P.bta[li] = Bb[li][2];
    P.btb[li] = Bb[li][3];
  }

  cfc_main<<<512 / BT, THREADS, 0, stream>>>(P);  // 32 workgroups
}
